// TransformerBlock_56538949484624
// MI455X (gfx1250) — matrix operand order repaired and checked by reading, not compiled, not run on hardware
//
#include <hip/hip_runtime.h>
#include <hip/hip_bf16.h>
#include <stdint.h>

// ---------------- types ----------------
typedef __bf16 bf16_t;
typedef __attribute__((ext_vector_type(16))) __bf16 v16bf;
typedef __attribute__((ext_vector_type(8)))  float  v8f;
typedef int v4i __attribute__((vector_size(16)));   // int32x4 for async-LDS builtins

#define AS1 __attribute__((address_space(1)))
#define AS3 __attribute__((address_space(3)))

#if __has_builtin(__builtin_amdgcn_global_load_async_to_lds_b128) && \
    __has_builtin(__builtin_amdgcn_s_wait_asynccnt)
#define USE_ASYNC_LDS 1
#else
#define USE_ASYNC_LDS 0
#endif

#define DIM     2048
#define SEQ     2048
#define NH      16
#define NKV     8
#define HD      128
#define HIDDEN  5632
#define QKV_N   4096   // (16 + 2*8) * 128
#define ATT_SCALE 0.08838834764831845f  // 1/sqrt(128)

__device__ __forceinline__ bf16_t f2bf(float f) {
  union { float f; uint32_t u; } a; a.f = f;
  uint32_t r = a.u + 0x7FFFu + ((a.u >> 16) & 1u);   // round-to-nearest-even
  union { uint16_t u; bf16_t b; } o; o.u = (uint16_t)(r >> 16);
  return o.b;
}

#if USE_ASYNC_LDS
// async 16B global -> LDS copy (ASYNCcnt-tracked, writes LDS directly)
__device__ __forceinline__ void async_cp16(AS3 void* dst, const void* src) {
  __builtin_amdgcn_global_load_async_to_lds_b128(
      (AS1 v4i*)src, (AS3 v4i*)dst, 0, 0);
}
#endif

union AFragU { v16bf v; uint32_t u[8]; };

// A-fragment (16x32 bf16, MxK): lane L: M=L&15; halves 0-7 -> K=kb..kb+7,
// halves 8-15 -> K=kb+16..kb+23, kb = k0 + (L<16 ? 0 : 8).  (ISA 7.12.2)
__device__ __forceinline__ v16bf load_a_frag(const bf16_t* base, int ld, int row0, int k0) {
  int lane = threadIdx.x & 31;
  int m  = lane & 15;
  int kb = k0 + ((lane < 16) ? 0 : 8);
  const bf16_t* p = base + (size_t)(row0 + m) * ld + kb;
  AFragU f;
  *(uint4*)&f.u[0] = *(const uint4*)(p);
  *(uint4*)&f.u[4] = *(const uint4*)(p + 16);
  return f.v;
}

// B-fragment (32x16 bf16, KxN), element (k,n) = base[(col0+n)*ld + k]:
// lane L: N=L&15; halves 0-7 -> K=kb..kb+7, halves 8-15 -> K=kb+16..kb+23, kb = k0 + (L<16 ? 0 : 8).
__device__ __forceinline__ v16bf load_b_frag(const bf16_t* base, int ld, int col0, int k0) {
  int lane = threadIdx.x & 31;
  int n  = lane & 15;
  int kb = k0 + ((lane < 16) ? 0 : 8);
  const bf16_t* p = base + (size_t)(col0 + n) * ld + kb;
  AFragU f;
  *(uint4*)&f.u[0] = *(const uint4*)(p);
  *(uint4*)&f.u[4] = *(const uint4*)(p + 16);
  return f.v;
}

__device__ __forceinline__ v8f wmma_bf16(v16bf a, v16bf b, v8f c) {
  return __builtin_amdgcn_wmma_f32_16x16x32_bf16(false, a, false, b, (short)0, c, false, false);
}

// ---------------- elementwise / norm kernels ----------------

__global__ __launch_bounds__(256) void cvt_f32_bf16(const float* __restrict__ in,
                                                    bf16_t* __restrict__ out, int n) {
  int i = blockIdx.x * 256 + threadIdx.x;
  if (i < n) out[i] = f2bf(in[i]);
}

// one row (DIM=2048) per block, 256 threads, 8 elems/thread
__global__ __launch_bounds__(256) void rmsnorm_bf16_kernel(const float* __restrict__ x,
                                                           const float* __restrict__ w,
                                                           bf16_t* __restrict__ out) {
  __shared__ float red[8];
  int s = blockIdx.x, tid = threadIdx.x;
  const float* row = x + (size_t)s * DIM;
  float v[8], ss = 0.f;
  #pragma unroll
  for (int i = 0; i < 8; i++) { v[i] = row[tid + i * 256]; ss += v[i] * v[i]; }
  #pragma unroll
  for (int off = 16; off; off >>= 1) ss += __shfl_xor(ss, off, 32);
  if ((tid & 31) == 0) red[tid >> 5] = ss;
  __syncthreads();
  float tot = 0.f;
  #pragma unroll
  for (int i = 0; i < 8; i++) tot += red[i];
  float sc = rsqrtf(tot / (float)DIM + 1e-5f);
  #pragma unroll
  for (int i = 0; i < 8; i++) {
    int c = tid + i * 256;
    out[(size_t)s * DIM + c] = f2bf(v[i] * sc * w[c]);
  }
}

// per-(token, head) LayerNorm + RoPE, bf16 outputs. grid=(SEQ, NH+2*NKV), block=128
__global__ __launch_bounds__(128) void qkv_post_kernel(const float* __restrict__ qkv,
                                                       const float* __restrict__ fc,
                                                       const float* __restrict__ fs,
                                                       const float* __restrict__ qw,
                                                       const float* __restrict__ qb,
                                                       const float* __restrict__ kw,
                                                       const float* __restrict__ kbias,
                                                       bf16_t* __restrict__ qo,
                                                       bf16_t* __restrict__ ko,
                                                       bf16_t* __restrict__ vo) {
  __shared__ float sh[HD];
  __shared__ float red[4];
  int s = blockIdx.x, j = blockIdx.y, d = threadIdx.x;
  int which; const float* src;
  if (j < NH)            { which = 0; src = qkv + (size_t)s * QKV_N + j * HD; }
  else if (j < NH + NKV) { which = 1; src = qkv + (size_t)s * QKV_N + NH * HD + (j - NH) * HD; }
  else                   { which = 2; src = qkv + (size_t)s * QKV_N + (NH + NKV) * HD + (j - NH - NKV) * HD; }
  float x = src[d];
  if (which == 2) {  // whole block takes this path -> no barrier divergence
    vo[((size_t)s * NKV + (j - NH - NKV)) * HD + d] = f2bf(x);
    return;
  }
  float t = x;
  #pragma unroll
  for (int off = 16; off; off >>= 1) t += __shfl_xor(t, off, 32);
  if ((d & 31) == 0) red[d >> 5] = t;
  __syncthreads();
  float mu = (red[0] + red[1] + red[2] + red[3]) * (1.f / HD);
  __syncthreads();
  float dv = (x - mu) * (x - mu);
  #pragma unroll
  for (int off = 16; off; off >>= 1) dv += __shfl_xor(dv, off, 32);
  if ((d & 31) == 0) red[d >> 5] = dv;
  __syncthreads();
  float var = (red[0] + red[1] + red[2] + red[3]) * (1.f / HD);
  float g = (which == 0) ? qw[d] : kw[d];
  float b = (which == 0) ? qb[d] : kbias[d];
  float ln = (x - mu) * rsqrtf(var + 1e-5f) * g + b;
  sh[d] = ln;
  __syncthreads();
  int p = d >> 1;
  float c = fc[(size_t)s * (HD / 2) + p], sn = fs[(size_t)s * (HD / 2) + p];
  float x0 = sh[2 * p], x1 = sh[2 * p + 1];
  float o = (d & 1) ? (x0 * sn + x1 * c) : (x0 * c - x1 * sn);
  if (which == 0) qo[((size_t)s * NH + j) * HD + d] = f2bf(o);
  else            ko[((size_t)s * NKV + (j - NH)) * HD + d] = f2bf(o);
}

// act = silu(x1) * x3, bf16 out
__global__ __launch_bounds__(256) void silu_gate_kernel(const float* __restrict__ x13,
                                                        bf16_t* __restrict__ act) {
  int i = blockIdx.x * 256 + threadIdx.x;
  int s = i / HIDDEN, c = i - s * HIDDEN;
  float a = x13[(size_t)s * (2 * HIDDEN) + c];
  float b = x13[(size_t)s * (2 * HIDDEN) + HIDDEN + c];
  float sig = 1.f / (1.f + __expf(-a));
  act[i] = f2bf(a * sig * b);
}

// ---------------- WMMA GEMM: C[M,N] = A[M,K] * W[N,K]^T (+resid) ----------------
#define BM 128
#define BN 128
#define BK 32
#if USE_ASYNC_LDS
#define NBUF 2
#else
#define NBUF 1
#endif

__global__ __launch_bounds__(256) void gemm_bf16_kernel(const bf16_t* __restrict__ A,
                                                        const bf16_t* __restrict__ W,
                                                        float* __restrict__ C,
                                                        const float* __restrict__ resid,
                                                        int M, int N, int K) {
  __shared__ __align__(16) bf16_t As[NBUF][BM][BK];
  __shared__ __align__(16) bf16_t Bs[NBUF][BN][BK];
  int tid = threadIdx.x;
  int wave  = tid >> 5;
  int waveM = wave & 3;   // 4 waves x 32 rows
  int waveN = wave >> 2;  // 2 waves x 64 cols
  int rowBase = blockIdx.y * BM;
  int colBase = blockIdx.x * BN;

  v8f acc[2][4];
  #pragma unroll
  for (int i = 0; i < 2; i++)
    #pragma unroll
    for (int j = 0; j < 4; j++)
      #pragma unroll
      for (int r = 0; r < 8; r++) acc[i][j][r] = 0.f;

  int lr = tid >> 1;            // 0..127
  int lc = (tid & 1) * 16;      // 0 / 16
  const bf16_t* Aptr = A + (size_t)(rowBase + lr) * K + lc;
  const bf16_t* Wptr = W + (size_t)(colBase + lr) * K + lc;
  int T = K / BK;

#if USE_ASYNC_LDS
  // software pipeline: ASYNCcnt-tracked global->LDS copies, double-buffered
  async_cp16((AS3 void*)&As[0][lr][lc],     Aptr);
  async_cp16((AS3 void*)&As[0][lr][lc + 8], Aptr + 8);
  async_cp16((AS3 void*)&Bs[0][lr][lc],     Wptr);
  async_cp16((AS3 void*)&Bs[0][lr][lc + 8], Wptr + 8);
  for (int t = 0; t < T; t++) {
    int cur = t & 1, nxt = cur ^ 1;
    if (t + 1 < T) {
      const bf16_t* An = Aptr + (size_t)(t + 1) * BK;
      const bf16_t* Wn = Wptr + (size_t)(t + 1) * BK;
      async_cp16((AS3 void*)&As[nxt][lr][lc],     An);
      async_cp16((AS3 void*)&As[nxt][lr][lc + 8], An + 8);
      async_cp16((AS3 void*)&Bs[nxt][lr][lc],     Wn);
      async_cp16((AS3 void*)&Bs[nxt][lr][lc + 8], Wn + 8);
      __builtin_amdgcn_s_wait_asynccnt(4);  // in-order: oldest 4 (tile t) done
    } else {
      __builtin_amdgcn_s_wait_asynccnt(0);
    }
    __syncthreads();
    v16bf af[2], bfr[4];
    #pragma unroll
    for (int i = 0; i < 2; i++) af[i] = load_a_frag(&As[cur][0][0], BK, waveM * 32 + i * 16, 0);
    #pragma unroll
    for (int j = 0; j < 4; j++) bfr[j] = load_b_frag(&Bs[cur][0][0], BK, waveN * 64 + j * 16, 0);
    #pragma unroll
    for (int i = 0; i < 2; i++)
      #pragma unroll
      for (int j = 0; j < 4; j++)
        acc[i][j] = wmma_bf16(af[i], bfr[j], acc[i][j]);
    __syncthreads();  // all waves done reading buf[cur] before t+2 overwrites it
  }
#else
  for (int t = 0; t < T; t++) {
    int k0 = t * BK;
    if (t + 1 < T) {          // next-tile prefetch -> global_prefetch_b8
      __builtin_prefetch(Aptr + k0 + BK, 0, 0);
      __builtin_prefetch(Wptr + k0 + BK, 0, 0);
    }
    uint4 a0 = *(const uint4*)(Aptr + k0);
    uint4 a1 = *(const uint4*)(Aptr + k0 + 8);
    uint4 b0 = *(const uint4*)(Wptr + k0);
    uint4 b1 = *(const uint4*)(Wptr + k0 + 8);
    *(uint4*)&As[0][lr][lc]     = a0;
    *(uint4*)&As[0][lr][lc + 8] = a1;
    *(uint4*)&Bs[0][lr][lc]     = b0;
    *(uint4*)&Bs[0][lr][lc + 8] = b1;
    __syncthreads();
    v16bf af[2], bfr[4];
    #pragma unroll
    for (int i = 0; i < 2; i++) af[i] = load_a_frag(&As[0][0][0], BK, waveM * 32 + i * 16, 0);
    #pragma unroll
    for (int j = 0; j < 4; j++) bfr[j] = load_b_frag(&Bs[0][0][0], BK, waveN * 64 + j * 16, 0);
    #pragma unroll
    for (int i = 0; i < 2; i++)
      #pragma unroll
      for (int j = 0; j < 4; j++)
        acc[i][j] = wmma_bf16(af[i], bfr[j], acc[i][j]);
    __syncthreads();
  }
#endif

  int lane = tid & 31;
  int n  = lane & 15;
  int mo = (lane < 16) ? 0 : 8;
  #pragma unroll
  for (int i = 0; i < 2; i++)
    #pragma unroll
    for (int j = 0; j < 4; j++) {
      int col = colBase + waveN * 64 + j * 16 + n;
      #pragma unroll
      for (int r = 0; r < 8; r++) {
        int row = rowBase + waveM * 32 + i * 16 + mo + r;
        float v = acc[i][j][r];
        if (resid) v += resid[(size_t)row * N + col];
        C[(size_t)row * N + col] = v;
      }
    }
}

// ---------------- flash attention (bf16 WMMA, online softmax) ----------------
// grid = (SEQ/64, NH), block = 128 (4 waves, 16 query rows each)
__global__ __launch_bounds__(128) void attn_kernel(const bf16_t* __restrict__ Q,
                                                   const bf16_t* __restrict__ Kc,
                                                   const bf16_t* __restrict__ Vc,
                                                   bf16_t* __restrict__ O) {
  __shared__ __align__(16) bf16_t Kb[32][HD];     // [key][d]
  __shared__ __align__(16) bf16_t Vt[HD][32];     // [d][key]  (transposed)
  __shared__ __align__(16) bf16_t Pw[4][16][32];  // per-wave P tile
  int head = blockIdx.y;
  int kvh  = head >> 1;                            // N_REP = 2
  int tid = threadIdx.x;
  int wave = tid >> 5, lane = tid & 31;
  int qBase = blockIdx.x * 64 + wave * 16;

  v16bf qf[4];
  #pragma unroll
  for (int c = 0; c < 4; c++) qf[c] = load_a_frag(Q + head * HD, NH * HD, qBase, c * 32);

  float m[8], l[8];
  v8f accO[8];
  #pragma unroll
  for (int r = 0; r < 8; r++) { m[r] = -INFINITY; l[r] = 0.f; }
  #pragma unroll
  for (int dt = 0; dt < 8; dt++)
    #pragma unroll
    for (int r = 0; r < 8; r++) accO[dt][r] = 0.f;

  int n  = lane & 15;
  int mo = (lane < 16) ? 0 : 8;
  int nkb = blockIdx.x * 2 + 2;   // causal: key blocks of 32 up to the diagonal

  for (int kb = 0; kb < nkb; kb++) {
    {
      int krow = tid >> 2;             // 0..31
      int kcol = (tid & 3) * 32;       // 32 halves per thread
      const bf16_t* kp = Kc + ((size_t)(kb * 32 + krow) * NKV + kvh) * HD + kcol;
      const bf16_t* vp = Vc + ((size_t)(kb * 32 + krow) * NKV + kvh) * HD + kcol;
#if USE_ASYNC_LDS
      async_cp16((AS3 void*)&Kb[krow][kcol],      kp);
      async_cp16((AS3 void*)&Kb[krow][kcol + 8],  kp + 8);
      async_cp16((AS3 void*)&Kb[krow][kcol + 16], kp + 16);
      async_cp16((AS3 void*)&Kb[krow][kcol + 24], kp + 24);
#else
      uint4 kv0 = ((const uint4*)kp)[0], kv1 = ((const uint4*)kp)[1];
      uint4 kv2 = ((const uint4*)kp)[2], kv3 = ((const uint4*)kp)[3];
      *(uint4*)&Kb[krow][kcol]      = kv0;
      *(uint4*)&Kb[krow][kcol + 8]  = kv1;
      *(uint4*)&Kb[krow][kcol + 16] = kv2;
      *(uint4*)&Kb[krow][kcol + 24] = kv3;
#endif
      bf16_t tmp[32];   // V must land transposed -> VALU/ds path
      ((uint4*)tmp)[0] = ((const uint4*)vp)[0];
      ((uint4*)tmp)[1] = ((const uint4*)vp)[1];
      ((uint4*)tmp)[2] = ((const uint4*)vp)[2];
      ((uint4*)tmp)[3] = ((const uint4*)vp)[3];
      #pragma unroll
      for (int e = 0; e < 32; e++) Vt[kcol + e][krow] = tmp[e];
    }
#if USE_ASYNC_LDS
    __builtin_amdgcn_s_wait_asynccnt(0);
#endif
    __syncthreads();

    // scores: S[16x32] = Q(16x128) x K^T(128x32)
    v8f s0, s1;
    #pragma unroll
    for (int r = 0; r < 8; r++) { s0[r] = 0.f; s1[r] = 0.f; }
    #pragma unroll
    for (int c = 0; c < 4; c++) {
      v16bf bk0 = load_b_frag(&Kb[0][0], HD, 0,  c * 32);
      v16bf bk1 = load_b_frag(&Kb[0][0], HD, 16, c * 32);
      s0 = wmma_bf16(qf[c], bk0, s0);
      s1 = wmma_bf16(qf[c], bk1, s1);
    }

    // online softmax (per row r; 16 cols live across the 16-lane half)
    float corr[8];
    #pragma unroll
    for (int r = 0; r < 8; r++) {
      int row = qBase + mo + r;
      int c0 = kb * 32 + n, c1 = c0 + 16;
      float a = s0[r] * ATT_SCALE + ((c0 > row) ? -1e9f : 0.f);
      float b = s1[r] * ATT_SCALE + ((c1 > row) ? -1e9f : 0.f);
      float mx = fmaxf(a, b);
      #pragma unroll
      for (int off = 8; off; off >>= 1) mx = fmaxf(mx, __shfl_xor(mx, off, 32));
      float mn = fmaxf(m[r], mx);
      corr[r] = __expf(m[r] - mn);
      m[r] = mn;
      float p0 = __expf(a - mn), p1 = __expf(b - mn);
      float sum = p0 + p1;
      #pragma unroll
      for (int off = 8; off; off >>= 1) sum += __shfl_xor(sum, off, 32);
      l[r] = l[r] * corr[r] + sum;
      Pw[wave][mo + r][n]      = f2bf(p0);
      Pw[wave][mo + r][16 + n] = f2bf(p1);
    }
    #pragma unroll
    for (int dt = 0; dt < 8; dt++)
      #pragma unroll
      for (int r = 0; r < 8; r++) accO[dt][r] *= corr[r];

    __builtin_amdgcn_wave_barrier();  // order per-wave LDS P writes vs reads

    // O += P(16x32) x V(32x128)
    v16bf pf = load_a_frag(&Pw[wave][0][0], 32, 0, 0);
    #pragma unroll
    for (int dt = 0; dt < 8; dt++) {
      v16bf bv = load_b_frag(&Vt[0][0], 32, dt * 16, 0);
      accO[dt] = wmma_bf16(pf, bv, accO[dt]);
    }
    __syncthreads();
  }

  #pragma unroll
  for (int dt = 0; dt < 8; dt++)
    #pragma unroll
    for (int r = 0; r < 8; r++) {
      int row = qBase + mo + r;
      float o = accO[dt][r] / l[r];
      O[(size_t)row * (NH * HD) + head * HD + dt * 16 + n] = f2bf(o);
    }
}

// ---------------- host orchestration ----------------
extern "C" void kernel_launch(void* const* d_in, const int* in_sizes, int n_in,
                              void* d_out, int out_size, void* d_ws, size_t ws_size,
                              hipStream_t stream) {
  const float* x       = (const float*)d_in[0];
  const float* fcos    = (const float*)d_in[1];
  const float* fsin    = (const float*)d_in[2];
  // d_in[3] = mask (causal mask computed in-kernel)
  const float* wqkv    = (const float*)d_in[4];
  const float* wo      = (const float*)d_in[5];
  const float* qn_w    = (const float*)d_in[6];
  const float* qn_b    = (const float*)d_in[7];
  const float* kn_w    = (const float*)d_in[8];
  const float* kn_b    = (const float*)d_in[9];
  const float* anorm_w = (const float*)d_in[10];
  const float* fnorm_w = (const float*)d_in[11];
  const float* w13     = (const float*)d_in[12];
  const float* w2      = (const float*)d_in[13];
  float* out = (float*)d_out;

  char* wsp = (char*)d_ws;
  auto alloc = [&](size_t bytes) { char* p = wsp; wsp += (bytes + 255) & ~(size_t)255; return p; };

  bf16_t* wqkv_bf = (bf16_t*)alloc((size_t)QKV_N * DIM * 2);
  bf16_t* wo_bf   = (bf16_t*)alloc((size_t)DIM * DIM * 2);
  bf16_t* w13_bf  = (bf16_t*)alloc((size_t)2 * HIDDEN * DIM * 2);
  bf16_t* w2_bf   = (bf16_t*)alloc((size_t)DIM * HIDDEN * 2);
  bf16_t* hbf     = (bf16_t*)alloc((size_t)SEQ * DIM * 2);
  float*  qkv32   = (float*) alloc((size_t)SEQ * QKV_N * 4);
  bf16_t* qbf     = (bf16_t*)alloc((size_t)SEQ * NH * HD * 2);
  bf16_t* kbf     = (bf16_t*)alloc((size_t)SEQ * NKV * HD * 2);
  bf16_t* vbf     = (bf16_t*)alloc((size_t)SEQ * NKV * HD * 2);
  bf16_t* obf     = (bf16_t*)alloc((size_t)SEQ * NH * HD * 2);
  float*  hres    = (float*) alloc((size_t)SEQ * DIM * 4);
  bf16_t* gbf     = (bf16_t*)alloc((size_t)SEQ * DIM * 2);
  float*  x13     = (float*) alloc((size_t)SEQ * 2 * HIDDEN * 4);
  bf16_t* actbf   = (bf16_t*)alloc((size_t)SEQ * HIDDEN * 2);
  (void)ws_size; (void)n_in; (void)in_sizes; (void)out_size;

  {
    int n1 = QKV_N * DIM, n2 = DIM * DIM, n3 = 2 * HIDDEN * DIM, n4 = DIM * HIDDEN;
    cvt_f32_bf16<<<(n1 + 255) / 256, 256, 0, stream>>>(wqkv, wqkv_bf, n1);
    cvt_f32_bf16<<<(n2 + 255) / 256, 256, 0, stream>>>(wo, wo_bf, n2);
    cvt_f32_bf16<<<(n3 + 255) / 256, 256, 0, stream>>>(w13, w13_bf, n3);
    cvt_f32_bf16<<<(n4 + 255) / 256, 256, 0, stream>>>(w2, w2_bf, n4);
  }
  rmsnorm_bf16_kernel<<<SEQ, 256, 0, stream>>>(x, anorm_w, hbf);
  gemm_bf16_kernel<<<dim3(QKV_N / BN, SEQ / BM), 256, 0, stream>>>(
      hbf, wqkv_bf, qkv32, nullptr, SEQ, QKV_N, DIM);
  qkv_post_kernel<<<dim3(SEQ, NH + 2 * NKV), 128, 0, stream>>>(
      qkv32, fcos, fsin, qn_w, qn_b, kn_w, kn_b, qbf, kbf, vbf);
  attn_kernel<<<dim3(SEQ / 64, NH), 128, 0, stream>>>(qbf, kbf, vbf, obf);
  gemm_bf16_kernel<<<dim3(DIM / BN, SEQ / BM), 256, 0, stream>>>(
      obf, wo_bf, hres, x, SEQ, DIM, DIM);
  rmsnorm_bf16_kernel<<<SEQ, 256, 0, stream>>>(hres, fnorm_w, gbf);
  gemm_bf16_kernel<<<dim3((2 * HIDDEN) / BN, SEQ / BM), 256, 0, stream>>>(
      gbf, w13_bf, x13, nullptr, SEQ, 2 * HIDDEN, DIM);
  {
    int n = SEQ * HIDDEN;
    silu_gate_kernel<<<(n + 255) / 256, 256, 0, stream>>>(x13, actbf);
  }
  gemm_bf16_kernel<<<dim3(DIM / BN, SEQ / BM), 256, 0, stream>>>(
      actbf, w2_bf, out, hres, SEQ, DIM, HIDDEN);
}
